// GCN_53695681135102
// MI455X (gfx1250) — matrix operand order repaired and checked by reading, not compiled, not run on hardware
//
#include <hip/hip_runtime.h>

// ---------------------------------------------------------------------------
// Types for CDNA5 WMMA (wave32): V_WMMA_F32_16X16X32_BF16
// ---------------------------------------------------------------------------
typedef __attribute__((ext_vector_type(16))) __bf16 bf16x16;
typedef __attribute__((ext_vector_type(8)))  float  f32x8;

__device__ __forceinline__ unsigned short f2bf(float f) {
    unsigned int u = __float_as_uint(f);
    unsigned int r = u + 0x7FFFu + ((u >> 16) & 1u);   // round-to-nearest-even
    return (unsigned short)(r >> 16);
}

// ---------------------------------------------------------------------------
// Pack x (f32 [N x 500]) -> bf16 A-fragments, K padded to 512.
// A-fragment layout (16-bit A 16x32, ISA 7.12.2):
//   element j of lane L maps to row = M0 + (L&15),
//   k_local = j + ((j>>3)<<3) + ((L>>4)<<3)
// Each thread emits 8 consecutive packed elements (one 16B store), whose
// source K values are contiguous: kbase = kt*32 + jhalf*16 + ((L>>4)<<3).
// ---------------------------------------------------------------------------
__global__ void pack_x_kernel(const float* __restrict__ x,
                              unsigned short* __restrict__ Ap, int MT) {
    const int numKT = 512 / 32;                  // 16
    int gid = blockIdx.x * blockDim.x + threadIdx.x;
    int total = MT * numKT * 64;                 // groups of 8 elements
    if (gid >= total) return;
    int jhalf = gid & 1;
    int lane  = (gid >> 1) & 31;
    int t     = gid >> 6;
    int kt    = t % numKT;
    int mt    = t / numKT;
    int rowi  = mt * 16 + (lane & 15);
    int kbase = kt * 32 + jhalf * 16 + ((lane >> 4) << 3);
    size_t po = ((size_t)(mt * numKT + kt) * 32 + lane) * 16 + jhalf * 8;
    const float* src = x + (size_t)rowi * 500;
#pragma unroll
    for (int jj = 0; jj < 8; ++jj) {
        int k = kbase + jj;
        float v = (k < 500) ? src[k] : 0.0f;
        Ap[po + jj] = f2bf(v);
    }
}

// ---------------------------------------------------------------------------
// Pack a weight matrix W (f32 [K0 x N0]) -> bf16 B-fragments [Kp x Np],
// zero-padded. B-fragment layout: element j of lane L:
//   k_local = (j&7) + ((j>>3)<<4) + ((L>>4)<<3), n_local = L & 15
// Packed index = ((kt*numNT + nt)*32 + lane)*16 + j
// ---------------------------------------------------------------------------
__global__ void pack_w_kernel(const float* __restrict__ W, int K0, int N0,
                              unsigned short* __restrict__ Bp, int Kp, int Np) {
    int idx = blockIdx.x * blockDim.x + threadIdx.x;
    int total = Kp * Np;
    if (idx >= total) return;
    int j    = idx & 15;
    int lane = (idx >> 4) & 31;
    int t    = idx >> 9;
    int numNT = Np / 16;
    int nt = t % numNT;
    int kt = t / numNT;
    int k = kt * 32 + (j & 7) + ((j >> 3) << 4) + ((lane >> 4) << 3);
    int n = nt * 16 + (lane & 15);
    float v = (k < K0 && n < N0) ? W[(size_t)k * N0 + n] : 0.0f;
    Bp[idx] = f2bf(v);
}

// ---------------------------------------------------------------------------
// GEMM: support[M x NT*16] (f32) = packedA (M x 32*numKT) * packedB.
// 8 waves/block; each wave owns one 16-row tile and keeps all NT output
// tiles in registers (A streams from HBM exactly once).
//
// B fragments are staged into LDS in chunks of KC=4 k-tiles (32 KB for NT=8),
// fetched from global once per BLOCK (not once per wave), then consumed via
// short-latency ds_load_b128. A has a 2-deep register prefetch. Register
// budget ~145 VGPRs (round-2's explicit global double-buffer spilled).
// ---------------------------------------------------------------------------
template <int NT>
__global__ void gemm_bf16_kernel(const unsigned short* __restrict__ Apu,
                                 const unsigned short* __restrict__ Bpu,
                                 float* __restrict__ C, int MT, int numKT) {
    constexpr int KC = 4;                          // k-tiles per LDS stage
    __shared__ unsigned short Bs[KC * NT * 512];   // NT=8: 32KB, NT=3: 12KB

    int wave = threadIdx.x >> 5;
    int lane = threadIdx.x & 31;
    int mt = blockIdx.x * 8 + wave;
    bool active = (mt < MT);                       // wave-uniform predicate

    const bf16x16* __restrict__ A = (const bf16x16*)Apu;
    const bf16x16* Arow = A + (size_t)mt * numKT * 32 + lane;
    const bf16x16* Bl = (const bf16x16*)Bs;

    f32x8 acc[NT] = {};
    bf16x16 a_cur = {}, a_nxt = {};
    if (active) a_cur = Arow[0];

    for (int kc = 0; kc < numKT; kc += KC) {
        __syncthreads();
        // Cooperative stage of B k-tiles [kc, kc+KC) into LDS (16B per copy).
        {
            const uint4* gB = (const uint4*)Bpu + (size_t)kc * NT * 64;
            uint4* sB = (uint4*)Bs;
            const int nchunks = KC * NT * 64;
            for (int i = threadIdx.x; i < nchunks; i += 256) sB[i] = gB[i];
        }
        __syncthreads();

        if (active) {
            for (int kk = 0; kk < KC; ++kk) {
                int kt = kc + kk;
                if (kt + 1 < numKT) a_nxt = Arow[(size_t)(kt + 1) * 32];
                bf16x16 b[NT];
#pragma unroll
                for (int nt = 0; nt < NT; ++nt)
                    b[nt] = Bl[(size_t)(kk * NT + nt) * 32 + lane];
#pragma unroll
                for (int nt = 0; nt < NT; ++nt)
                    acc[nt] = __builtin_amdgcn_wmma_f32_16x16x32_bf16(
                        false, a_cur, false, b[nt], (short)0, acc[nt], false, false);
                a_cur = a_nxt;
            }
        }
    }

    if (!active) return;

    // C/D layout: VGPR r -> row r + ((lane>>4)<<3), col = lane&15
    const int ldc = NT * 16;
    int r0 = (lane >> 4) << 3;
    int cN = lane & 15;
#pragma unroll
    for (int nt = 0; nt < NT; ++nt) {
#pragma unroll
        for (int r = 0; r < 8; ++r) {
            C[(size_t)(mt * 16 + r0 + r) * ldc + nt * 16 + cN] = acc[nt][r];
        }
    }
}

// ---------------------------------------------------------------------------
// Zero fill (float4 granularity)
// ---------------------------------------------------------------------------
__global__ void zero_f32_kernel(float* __restrict__ p, int n4) {
    int i = blockIdx.x * blockDim.x + threadIdx.x;
    if (i >= n4) return;
    float4* q = (float4*)p;
    q[i] = make_float4(0.f, 0.f, 0.f, 0.f);
}

// ---------------------------------------------------------------------------
// SpMM scatter: agg[row[e]] += ew[e] * support[col[e]]  (f32 atomics, L2-hot)
// One thread per (edge, 4 features).
// ---------------------------------------------------------------------------
__global__ void spmm_kernel(const int* __restrict__ row, const int* __restrict__ col,
                            const float* __restrict__ ew,
                            const float* __restrict__ S, float* __restrict__ agg,
                            int E, int Nc) {
    int per = Nc >> 2;
    long gid = (long)blockIdx.x * blockDim.x + threadIdx.x;
    if (gid >= (long)E * per) return;
    int e = (int)(gid / per);
    int f = (int)(gid % per) * 4;
    float w = ew[e];
    int c = col[e], rdst = row[e];
    const float4 s = *(const float4*)(S + (size_t)c * Nc + f);
    float* dst = agg + (size_t)rdst * Nc + f;
    atomicAdd(dst + 0, w * s.x);
    atomicAdd(dst + 1, w * s.y);
    atomicAdd(dst + 2, w * s.z);
    atomicAdd(dst + 3, w * s.w);
}

// ---------------------------------------------------------------------------
// Bias + ReLU fused with bf16 A-fragment repack for the next layer (K = 128).
// ---------------------------------------------------------------------------
__global__ void finalize_pack_kernel(const float* __restrict__ agg,
                                     const float* __restrict__ bias,
                                     unsigned short* __restrict__ Ap,
                                     int MT, int K) {
    int numKT = K / 32;
    int gid = blockIdx.x * blockDim.x + threadIdx.x;
    int total = MT * numKT * 64;
    if (gid >= total) return;
    int jhalf = gid & 1;
    int lane  = (gid >> 1) & 31;
    int t     = gid >> 6;
    int kt    = t % numKT;
    int mt    = t / numKT;
    int rowi  = mt * 16 + (lane & 15);
    int kbase = kt * 32 + jhalf * 16 + ((lane >> 4) << 3);
    size_t po = ((size_t)(mt * numKT + kt) * 32 + lane) * 16 + jhalf * 8;
    const float* src = agg + (size_t)rowi * K;
#pragma unroll
    for (int jj = 0; jj < 8; ++jj) {
        int k = kbase + jj;
        float v = src[k] + bias[k];
        v = fmaxf(v, 0.0f);
        Ap[po + jj] = f2bf(v);
    }
}

// ---------------------------------------------------------------------------
// Final layer: bias add + log_softmax over 40 classes (agg stride 48).
// ---------------------------------------------------------------------------
__global__ void logsoftmax_kernel(const float* __restrict__ agg,
                                  const float* __restrict__ b,
                                  float* __restrict__ out, int M) {
    int r = blockIdx.x * blockDim.x + threadIdx.x;
    if (r >= M) return;
    const float* a = agg + (size_t)r * 48;
    float v[40];
    float mx = -3.4e38f;
#pragma unroll
    for (int i = 0; i < 40; ++i) { v[i] = a[i] + b[i]; mx = fmaxf(mx, v[i]); }
    float s = 0.0f;
#pragma unroll
    for (int i = 0; i < 40; ++i) s += expf(v[i] - mx);
    float ls = logf(s);
    float* o = out + (size_t)r * 40;
#pragma unroll
    for (int i = 0; i < 40; ++i) o[i] = v[i] - mx - ls;
}

// ---------------------------------------------------------------------------
// Host orchestration
// ---------------------------------------------------------------------------
static inline size_t alignup256(size_t x) { return (x + 255) & ~(size_t)255; }

extern "C" void kernel_launch(void* const* d_in, const int* in_sizes, int n_in,
                              void* d_out, int out_size, void* d_ws, size_t ws_size,
                              hipStream_t stream) {
    const float* x   = (const float*)d_in[0];
    const int*   row = (const int*)d_in[1];
    const int*   col = (const int*)d_in[2];
    const float* ew  = (const float*)d_in[3];
    const float* W1  = (const float*)d_in[4];
    const float* b1  = (const float*)d_in[5];
    const float* W2  = (const float*)d_in[6];
    const float* b2  = (const float*)d_in[7];
    const float* W3  = (const float*)d_in[8];
    const float* b3  = (const float*)d_in[9];
    const float* W4  = (const float*)d_in[10];
    const float* b4  = (const float*)d_in[11];
    float* out = (float*)d_out;

    const int N  = in_sizes[0] / 500;   // 100000
    const int E  = in_sizes[1];         // 1600000
    const int MT = N / 16;              // 6250 row tiles

    // ---- workspace carve-up ----
    char* ws = (char*)d_ws;
    size_t off = 0;
    unsigned short* A1 = (unsigned short*)(ws + off); off = alignup256(off + (size_t)N * 512 * 2); // packed x / packed h2
    unsigned short* A2 = (unsigned short*)(ws + off); off = alignup256(off + (size_t)N * 128 * 2); // packed h1 / packed h3
    float* supp = (float*)(ws + off); off = alignup256(off + (size_t)N * 128 * 4);
    float* agg  = (float*)(ws + off); off = alignup256(off + (size_t)N * 128 * 4);
    unsigned short* B1p = (unsigned short*)(ws + off); off = alignup256(off + (size_t)512 * 128 * 2);
    unsigned short* B2p = (unsigned short*)(ws + off); off = alignup256(off + (size_t)128 * 128 * 2);
    unsigned short* B3p = (unsigned short*)(ws + off); off = alignup256(off + (size_t)128 * 128 * 2);
    unsigned short* B4p = (unsigned short*)(ws + off); off = alignup256(off + (size_t)128 * 48 * 2);

    const dim3 blk(256);
    auto gct = [](long n) { return dim3((unsigned)((n + 255) / 256)); };

    // ---- one-time packing (deterministic, every call) ----
    pack_w_kernel<<<gct(512 * 128), blk, 0, stream>>>(W1, 500, 128, B1p, 512, 128);
    pack_w_kernel<<<gct(128 * 128), blk, 0, stream>>>(W2, 128, 128, B2p, 128, 128);
    pack_w_kernel<<<gct(128 * 128), blk, 0, stream>>>(W3, 128, 128, B3p, 128, 128);
    pack_w_kernel<<<gct(128 * 48),  blk, 0, stream>>>(W4, 128, 40,  B4p, 128, 48);
    pack_x_kernel<<<gct((long)MT * 1024), blk, 0, stream>>>(x, A1, MT);

    const dim3 ggemm((MT + 7) / 8);
    const long spmm128 = (long)E * 32;   // E * (128/4)
    const long spmm48  = (long)E * 12;   // E * (48/4)

    // ---- layer 1: x @ W1 -> spmm -> relu -> packed A2 ----
    gemm_bf16_kernel<8><<<ggemm, blk, 0, stream>>>(A1, B1p, supp, MT, 16);
    zero_f32_kernel<<<gct((long)N * 32), blk, 0, stream>>>(agg, N * 32);
    spmm_kernel<<<gct(spmm128), blk, 0, stream>>>(row, col, ew, supp, agg, E, 128);
    finalize_pack_kernel<<<gct((long)MT * 256), blk, 0, stream>>>(agg, b1, A2, MT, 128);

    // ---- layer 2: h1 @ W2 -> spmm -> relu -> packed A1 (reuse) ----
    gemm_bf16_kernel<8><<<ggemm, blk, 0, stream>>>(A2, B2p, supp, MT, 4);
    zero_f32_kernel<<<gct((long)N * 32), blk, 0, stream>>>(agg, N * 32);
    spmm_kernel<<<gct(spmm128), blk, 0, stream>>>(row, col, ew, supp, agg, E, 128);
    finalize_pack_kernel<<<gct((long)MT * 256), blk, 0, stream>>>(agg, b2, A1, MT, 128);

    // ---- layer 3: h2 @ W3 -> spmm -> relu -> packed A2 (reuse) ----
    gemm_bf16_kernel<8><<<ggemm, blk, 0, stream>>>(A1, B3p, supp, MT, 4);
    zero_f32_kernel<<<gct((long)N * 32), blk, 0, stream>>>(agg, N * 32);
    spmm_kernel<<<gct(spmm128), blk, 0, stream>>>(row, col, ew, supp, agg, E, 128);
    finalize_pack_kernel<<<gct((long)MT * 256), blk, 0, stream>>>(agg, b3, A2, MT, 128);

    // ---- layer 4: h3 @ W4 (N padded 40->48) -> spmm -> log_softmax ----
    gemm_bf16_kernel<3><<<ggemm, blk, 0, stream>>>(A2, B4p, supp, MT, 4);
    zero_f32_kernel<<<gct((long)N * 12), blk, 0, stream>>>(agg, N * 12);
    spmm_kernel<<<gct(spmm48), blk, 0, stream>>>(row, col, ew, supp, agg, E, 48);
    logsoftmax_kernel<<<gct(N), blk, 0, stream>>>(agg, b4, out, N);
}
